// LinearAttention_27917287424011
// MI455X (gfx1250) — matrix operand order repaired and checked by reading, not compiled, not run on hardware
//
#include <hip/hip_runtime.h>
#include <hip/hip_bf16.h>
#include <cstdint>

typedef __bf16 bf16;
typedef __attribute__((ext_vector_type(16))) __bf16 v16bf;
typedef __attribute__((ext_vector_type(8)))  __bf16 v8bf;
typedef __attribute__((ext_vector_type(8)))  float  v8f;

// Problem constants: B=4, NQ=NK=8192, D=512, H=8, HD=64  -> 32768 token rows.

__device__ __forceinline__ v16bf frag2(const bf16* p0, const bf16* p1) {
  union { v16bf v; v8bf h[2]; } u;
  u.h[0] = *(const v8bf*)p0;   // 16B -> ds_load_b128 / global_load_b128
  u.h[1] = *(const v8bf*)p1;
  return u.v;
}

__device__ __forceinline__ v8f wmma_bf16(v16bf a, v16bf b, v8f c) {
  return __builtin_amdgcn_wmma_f32_16x16x32_bf16(false, a, false, b, (short)0, c,
                                                 false, false);
}

__device__ __forceinline__ void gatomic_add_f32(float* p, float v) {
  asm volatile("global_atomic_add_f32 %0, %1, off" :: "v"(p), "v"(v) : "memory");
}

// ---------------------------------------------------------------- weights->bf16
__global__ __launch_bounds__(256) void prep_weights(
    const float* Wq, const float* Wk, const float* Wv, const float* Wo,
    bf16* wq, bf16* wk, bf16* wv, bf16* wo) {
  int t = blockIdx.x * 256 + threadIdx.x;     // 262144 threads, 1 elem per matrix
  if (t < 512 * 512) {
    wq[t] = (bf16)Wq[t];
    wk[t] = (bf16)Wk[t];
    wv[t] = (bf16)Wv[t];
    wo[t] = (bf16)Wo[t];
  }
}

// ------------------------------------------------- fused QKV projection + featmap
// 128x64 tile / block, 8 waves, each wave: 16 rows x 64 cols (4 WMMA / K-step).
// Double-buffered LDS A staging -> one barrier per K-step.
__global__ __launch_bounds__(256) void proj_kernel(
    const float* Xq, const float* Xk, const float* Xv,
    const bf16* Wqb, const bf16* Wkb, const bf16* Wvb,
    const float* bq, const float* bk, const float* bv,
    bf16* Oq, bf16* Ok, bf16* Ov) {
  __shared__ bf16 As[2][128 * 40];             // 128 rows x 32 K, pad to 40
  const float* X; const bf16* W; const float* bias; bf16* O; int fm;
  if (blockIdx.z == 0)      { X = Xq; W = Wqb; bias = bq; O = Oq; fm = 1; }
  else if (blockIdx.z == 1) { X = Xk; W = Wkb; bias = bk; O = Ok; fm = 1; }
  else                      { X = Xv; W = Wvb; bias = bv; O = Ov; fm = 0; }

  const int t = threadIdx.x;
  const int mBase = blockIdx.x * 128, nBase = blockIdx.y * 64;
  const int wid = t >> 5, lane = t & 31, lo = lane & 15, hi = lane >> 4;
  const int srow = t >> 1, scol = (t & 1) * 16;   // stage: 16 elems / thread
  const float* srcRow = X + (size_t)(mBase + srow) * 512 + scol;

  // prologue: stage K-chunk 0 into buffer 0
  {
    const float4* src = (const float4*)srcRow;
    float4 f0 = src[0], f1 = src[1], f2 = src[2], f3 = src[3];
    v8bf a0 = { (bf16)f0.x, (bf16)f0.y, (bf16)f0.z, (bf16)f0.w,
                (bf16)f1.x, (bf16)f1.y, (bf16)f1.z, (bf16)f1.w };
    v8bf a1 = { (bf16)f2.x, (bf16)f2.y, (bf16)f2.z, (bf16)f2.w,
                (bf16)f3.x, (bf16)f3.y, (bf16)f3.z, (bf16)f3.w };
    *(v8bf*)&As[0][srow * 40 + scol]     = a0;
    *(v8bf*)&As[0][srow * 40 + scol + 8] = a1;
  }

  v8f acc[4] = {v8f{}, v8f{}, v8f{}, v8f{}};
  for (int it = 0; it < 16; ++it) {
    __syncthreads();                           // staging of buf[it&1] complete
    const int k0 = it * 32;
    if (it < 15) {                             // stage next chunk into other buf
      const float4* src = (const float4*)(srcRow + k0 + 32);
      float4 f0 = src[0], f1 = src[1], f2 = src[2], f3 = src[3];
      v8bf a0 = { (bf16)f0.x, (bf16)f0.y, (bf16)f0.z, (bf16)f0.w,
                  (bf16)f1.x, (bf16)f1.y, (bf16)f1.z, (bf16)f1.w };
      v8bf a1 = { (bf16)f2.x, (bf16)f2.y, (bf16)f2.z, (bf16)f2.w,
                  (bf16)f3.x, (bf16)f3.y, (bf16)f3.z, (bf16)f3.w };
      const int nb = (it + 1) & 1;
      *(v8bf*)&As[nb][srow * 40 + scol]     = a0;
      *(v8bf*)&As[nb][srow * 40 + scol + 8] = a1;
    }
    const bf16* arow = &As[it & 1][(wid * 16 + lo) * 40];
    v16bf af = frag2(arow + hi * 8, arow + 16 + hi * 8);
    #pragma unroll
    for (int j = 0; j < 4; ++j) {
      const bf16* wp = W + (size_t)(nBase + j * 16 + lo) * 512 + k0;
      acc[j] = wmma_bf16(af, frag2(wp + hi * 8, wp + 16 + hi * 8), acc[j]);
    }
  }
  #pragma unroll
  for (int j = 0; j < 4; ++j) {
    int n = nBase + j * 16 + lo;
    float bb = bias[n];
    #pragma unroll
    for (int r = 0; r < 8; ++r) {
      int m = mBase + wid * 16 + r + 8 * hi;
      float v = acc[j][r] + bb;
      if (fm) v = (v > 0.f) ? (v + 1.f) : __expf(v);
      O[(size_t)m * 512 + n] = (bf16)v;
    }
  }
}

// -------------------------------------------- kv = k^T v  (64x64 / head), k_sum
__global__ __launch_bounds__(256) void kv_kernel(
    const bf16* Kf, const bf16* Vf, float* kv_g, float* ksum_g) {
  __shared__ bf16 kT[64 * 40];                 // [d][n] transposed
  __shared__ bf16 vT[64 * 40];                 // [m][n] transposed
  __shared__ float ksum_s[64];
  const int t = threadIdx.x;
  const int bh = blockIdx.y, b = bh >> 3, h = bh & 7;
  const int ns = blockIdx.x * 1024;            // 8 splits x 1024 rows
  const int nl = t >> 3, e0 = (t & 7) * 8;
  const int wid = t >> 5, lane = t & 31, lo = lane & 15, hi = lane >> 4;
  const int dt = wid >> 1, mp = wid & 1;

  float ks[8] = {};
  v8f acc0 = {}, acc1 = {};
  for (int it = 0; it < 32; ++it) {
    int n = ns + it * 32 + nl;
    size_t rbase = ((size_t)(b * 8192 + n)) * 512 + h * 64 + e0;
    v8bf kk = *(const v8bf*)(Kf + rbase);
    v8bf vv = *(const v8bf*)(Vf + rbase);
    #pragma unroll
    for (int i = 0; i < 8; ++i) {
      ks[i] += (float)kk[i];
      kT[(e0 + i) * 40 + nl] = kk[i];
      vT[(e0 + i) * 40 + nl] = vv[i];
    }
    __syncthreads();
    const bf16* arow = &kT[(dt * 16 + lo) * 40];
    v16bf af = frag2(arow + hi * 8, arow + 16 + hi * 8);
    const bf16* b0 = &vT[(mp * 32 + lo) * 40];
    const bf16* b1 = &vT[(mp * 32 + 16 + lo) * 40];
    acc0 = wmma_bf16(af, frag2(b0 + hi * 8, b0 + 16 + hi * 8), acc0);
    acc1 = wmma_bf16(af, frag2(b1 + hi * 8, b1 + 16 + hi * 8), acc1);
    __syncthreads();
  }
  float* kvb = kv_g + (size_t)bh * 4096;
  for (int j = 0; j < 2; ++j) {
    v8f acc = j ? acc1 : acc0;
    int m = mp * 32 + j * 16 + lo;
    #pragma unroll
    for (int r = 0; r < 8; ++r) {
      int d = dt * 16 + r + 8 * hi;
      gatomic_add_f32(kvb + d * 64 + m, acc[r]);
    }
  }
  if (t < 64) ksum_s[t] = 0.f;
  __syncthreads();
  #pragma unroll
  for (int i = 0; i < 8; ++i) atomicAdd(&ksum_s[e0 + i], ks[i]);  // ds_add_f32
  __syncthreads();
  if (t < 64) gatomic_add_f32(ksum_g + (size_t)bh * 64 + t, ksum_s[t]);
}

// ---------------------------------- out = (q . kv) * 1/(q.k_sum + 1e-6), bf16 out
__global__ __launch_bounds__(256) void attn_kernel(
    const bf16* Qf, const float* kv_g, const float* ksum_g, bf16* At) {
  __shared__ bf16 kvT[8][64 * 80];             // per-head kv^T [m][d], pad 80
  __shared__ float ksum_s[512];
  __shared__ float z_s[512];
  const int t = threadIdx.x;
  const int rowBase = blockIdx.x * 64;
  const int b = rowBase >> 13;                 // /8192
  const int wid = t >> 5, lane = t & 31, lo = lane & 15, hi = lane >> 4;
  const int h = wid;                           // wave == head
  const float* kvb = kv_g + ((size_t)b * 8 + h) * 4096;
  #pragma unroll 4
  for (int i = 0; i < 128; ++i) {
    int idx = i * 32 + lane;                   // coalesced read, scattered LDS write
    kvT[h][(idx & 63) * 80 + (idx >> 6)] = (bf16)kvb[idx];
  }
  for (int i = t; i < 512; i += 256) ksum_s[i] = ksum_g[(size_t)b * 512 + i];
  __syncthreads();
  for (int idx = t; idx < 512; idx += 256) {
    int rl = idx & 63, hh = idx >> 6;
    const bf16* qr = Qf + (size_t)(rowBase + rl) * 512 + hh * 64;
    float dot = 0.f;
    for (int d = 0; d < 64; ++d) dot += (float)qr[d] * ksum_s[hh * 64 + d];
    z_s[idx] = 1.f / (dot + 1e-6f);
  }
  __syncthreads();
  const bf16* kvh = kvT[h];
  for (int nsub = 0; nsub < 4; ++nsub) {
    const bf16* qr = Qf + (size_t)(rowBase + nsub * 16 + lo) * 512 + h * 64;
    v16bf a0 = frag2(qr + hi * 8,      qr + 16 + hi * 8);   // K = 0..31
    v16bf a1 = frag2(qr + 32 + hi * 8, qr + 48 + hi * 8);   // K = 32..63
    for (int msub = 0; msub < 4; ++msub) {
      const bf16* bp = &kvh[(msub * 16 + lo) * 80];
      v8f acc = {};
      acc = wmma_bf16(a0, frag2(bp + hi * 8,      bp + 16 + hi * 8),  acc);
      acc = wmma_bf16(a1, frag2(bp + 32 + hi * 8, bp + 48 + hi * 8), acc);
      #pragma unroll
      for (int r = 0; r < 8; ++r) {
        int rl = nsub * 16 + r + 8 * hi;
        float v = acc[r] * z_s[h * 64 + rl];
        At[(size_t)(rowBase + rl) * 512 + h * 64 + msub * 16 + lo] = (bf16)v;
      }
    }
  }
}

// ------------------------------------------------- output projection (fp32 out)
// Same 128x64 structure as proj_kernel, bf16 input staging, fp32 output.
__global__ __launch_bounds__(256) void outproj_kernel(
    const bf16* At, const bf16* Wob, const float* bo, float* Out) {
  __shared__ bf16 As[2][128 * 40];
  const int t = threadIdx.x;
  const int mBase = blockIdx.x * 128, nBase = blockIdx.y * 64;
  const int wid = t >> 5, lane = t & 31, lo = lane & 15, hi = lane >> 4;
  const int srow = t >> 1, scol = (t & 1) * 16;
  const bf16* srcRow = At + (size_t)(mBase + srow) * 512 + scol;

  {
    const v8bf* src = (const v8bf*)srcRow;
    *(v8bf*)&As[0][srow * 40 + scol]     = src[0];
    *(v8bf*)&As[0][srow * 40 + scol + 8] = src[1];
  }
  v8f acc[4] = {v8f{}, v8f{}, v8f{}, v8f{}};
  for (int it = 0; it < 16; ++it) {
    __syncthreads();
    const int k0 = it * 32;
    if (it < 15) {
      const v8bf* src = (const v8bf*)(srcRow + k0 + 32);
      const int nb = (it + 1) & 1;
      *(v8bf*)&As[nb][srow * 40 + scol]     = src[0];
      *(v8bf*)&As[nb][srow * 40 + scol + 8] = src[1];
    }
    const bf16* arow = &As[it & 1][(wid * 16 + lo) * 40];
    v16bf af = frag2(arow + hi * 8, arow + 16 + hi * 8);
    #pragma unroll
    for (int j = 0; j < 4; ++j) {
      const bf16* wp = Wob + (size_t)(nBase + j * 16 + lo) * 512 + k0;
      acc[j] = wmma_bf16(af, frag2(wp + hi * 8, wp + 16 + hi * 8), acc[j]);
    }
  }
  #pragma unroll
  for (int j = 0; j < 4; ++j) {
    int n = nBase + j * 16 + lo;
    float bb = bo[n];
    #pragma unroll
    for (int r = 0; r < 8; ++r) {
      int m = mBase + wid * 16 + r + 8 * hi;
      Out[(size_t)m * 512 + n] = acc[j][r] + bb;
    }
  }
}

// ---------------------------------------------------------------------- launch
extern "C" void kernel_launch(void* const* d_in, const int* in_sizes, int n_in,
                              void* d_out, int out_size, void* d_ws, size_t ws_size,
                              hipStream_t stream) {
  const float* query = (const float*)d_in[0];
  const float* key_  = (const float*)d_in[1];
  const float* value = (const float*)d_in[2];
  const float* Wq = (const float*)d_in[3];
  const float* bq = (const float*)d_in[4];
  const float* Wk = (const float*)d_in[5];
  const float* bk = (const float*)d_in[6];
  const float* Wv = (const float*)d_in[7];
  const float* bv = (const float*)d_in[8];
  const float* Wo = (const float*)d_in[9];
  const float* bo = (const float*)d_in[10];

  char* ws = (char*)d_ws;
  bf16*  wq_b = (bf16*)(ws + 0);               //  512 KB
  bf16*  wk_b = (bf16*)(ws + 524288);          //  512 KB
  bf16*  wv_b = (bf16*)(ws + 1048576);         //  512 KB
  bf16*  wo_b = (bf16*)(ws + 1572864);         //  512 KB
  bf16*  qf   = (bf16*)(ws + 2097152);         //   32 MB
  bf16*  kf   = (bf16*)(ws + 35651584);        //   32 MB
  bf16*  vf   = (bf16*)(ws + 69206016);        //   32 MB
  bf16*  at   = (bf16*)(ws + 102760448);       //   32 MB
  float* kv   = (float*)(ws + 136314880);      //  512 KB
  float* ksum = (float*)(ws + 136839168);      //    8 KB

  hipMemsetAsync(kv, 0, 524288 + 8192, stream);          // kv + ksum contiguous
  prep_weights<<<1024, 256, 0, stream>>>(Wq, Wk, Wv, Wo, wq_b, wk_b, wv_b, wo_b);
  proj_kernel<<<dim3(256, 8, 3), 256, 0, stream>>>(
      query, key_, value, wq_b, wk_b, wv_b, bq, bk, bv, qf, kf, vf);
  kv_kernel<<<dim3(8, 32), 256, 0, stream>>>(kf, vf, kv, ksum);
  attn_kernel<<<512, 256, 0, stream>>>(qf, kv, ksum, at);
  outproj_kernel<<<dim3(256, 8), 256, 0, stream>>>(at, wo_b, bo, (float*)d_out);
}
